// SubsetsSampleWeighted_71347996721713
// MI455X (gfx1250) — matrix operand order repaired and checked by reading, not compiled, not run on hardware
//
#include <hip/hip_runtime.h>
#include <hip/hip_bf16.h>

typedef __attribute__((ext_vector_type(16))) _Float16 v16h;
typedef __attribute__((ext_vector_type(8)))  _Float16 v8h;
typedef __attribute__((ext_vector_type(4)))  _Float16 v4h;
typedef __attribute__((ext_vector_type(8)))  float    v8f;

#define NB 16
#define NA 64
#define NG 128
#define NS 8192
#define NM 16
#define ND 128
#define SPECT_BIN_N 512
#define LN_EPS 1e-5f

// ---------------- WMMA helpers (gfx1250, wave32) ----------------

__device__ __forceinline__ v8f wmma16(v16h a, v16h b, v8f c) {
  // D = A(16x32 f16) * B(32x16 f16) + C(16x16 f32)
  return __builtin_amdgcn_wmma_f32_16x16x32_f16(
      /*neg_a=*/false, a, /*neg_b=*/false, b,
      /*c_mod=*/(short)0, c, /*reuse_a=*/false, /*reuse_b=*/false);
}

// A fragment: 16x32 f16 tile from row-major [row][K] LDS buffer (stride ldk halves).
// ISA layout: lanes 0-15: elems0-7 = K0..7,  elems8-15 = K16..23
//             lanes16-31: elems0-7 = K8..15, elems8-15 = K24..31
__device__ __forceinline__ v16h frag_a(const _Float16* base, int row, int ldk,
                                       int k0, int lane) {
  const _Float16* p = base + row * ldk + k0 + ((lane >> 4) & 1) * 8;
  v8h lo = *(const v8h*)p;
  v8h hi = *(const v8h*)(p + 16);
  v16h r;
#pragma unroll
  for (int i = 0; i < 8; ++i) { r[i] = lo[i]; r[i + 8] = hi[i]; }
  return r;
}

// B fragment: 32x16 f16 tile from N-major [col][K] LDS buffer (stride ldk halves).
// ISA layout: lanes 0-15: elems0-7 = K0..7, elems8-15 = K16..23; lanes 16-31: K8..15, K24..31; N = lane&15.
__device__ __forceinline__ v16h frag_b(const _Float16* baseT, int col, int ldk,
                                       int k0, int lane) {
  const _Float16* p = baseT + col * ldk + k0 + ((lane >> 4) & 1) * 8;
  v8h lo = *(const v8h*)p;
  v8h hi = *(const v8h*)(p + 16);
  v16h r;
#pragma unroll
  for (int i = 0; i < 8; ++i) { r[i] = lo[i]; r[i + 8] = hi[i]; }
  return r;
}

// Sum across the 16-lane half-wave group holding one output row.
__device__ __forceinline__ float red16(float v) {
  v += __shfl_xor(v, 1);
  v += __shfl_xor(v, 2);
  v += __shfl_xor(v, 4);
  v += __shfl_xor(v, 8);
  return v;
}

// ---------------- Kernel 1: fused pool + MLP scorer ----------------
// grid: (NS/128, NB), block: 256 (8 waves); each wave owns a 16-row strip.

__global__ __launch_bounds__(256) void fused_scorer(
    const float* __restrict__ feats,   // (B,A,G)
    const float* __restrict__ mask,    // (B,A)
    const int*   __restrict__ subsets, // (B,S,A)
    const float* __restrict__ ln0w, const float* __restrict__ ln0b,
    const float* __restrict__ W1g, const float* __restrict__ b1g,
    const float* __restrict__ W2g, const float* __restrict__ b2g,
    const float* __restrict__ ln1w, const float* __restrict__ ln1b,
    const float* __restrict__ Wsg, const float* __restrict__ bsg,
    float* __restrict__ scores)        // (B,S)
{
  __shared__ __align__(32) _Float16 sA [128 * 64];   // subs tile   [row][a]
  __shared__ __align__(32) _Float16 sBt[128 * 64];   // masked feat [g][a]
  __shared__ __align__(32) _Float16 sX [128 * 128];  // activations [row][k]
  __shared__ __align__(32) _Float16 sH [128 * 128];  // activations [row][k]
  __shared__ __align__(32) _Float16 sW [128 * 128];  // weights     [n][k]
  __shared__ float sMask[64];
  __shared__ float sSize[128];
  __shared__ float sP[7 * 128];  // ln0w ln0b b1 b2 ln1w ln1b Ws
  __shared__ float sBs;

  const int tid = threadIdx.x;
  const int b   = blockIdx.y;
  const int s0  = blockIdx.x * 128;

  if (tid < 64) sMask[tid] = mask[b * NA + tid];
  if (tid < 128) {
    sP[0 * 128 + tid] = ln0w[tid];
    sP[1 * 128 + tid] = ln0b[tid];
    sP[2 * 128 + tid] = b1g[tid];
    sP[3 * 128 + tid] = b2g[tid];
    sP[4 * 128 + tid] = ln1w[tid];
    sP[5 * 128 + tid] = ln1b[tid];
    sP[6 * 128 + tid] = Wsg[tid];
  }
  if (tid == 0) sBs = bsg[0];
  __syncthreads();

  // masked features -> sBt[g][a]; float4 over contiguous g, scatter 4 f16 stores
  {
    const float4* f4 = (const float4*)(feats + (size_t)b * NA * NG);
    for (int i = tid; i < NA * (NG / 4); i += 256) {   // 2048 chunks
      int a = i >> 5, g4 = i & 31;
      float4 v = f4[a * (NG / 4) + g4];
      float m = sMask[a];
      int g = g4 * 4;
      sBt[(g + 0) * 64 + a] = (_Float16)(v.x * m);
      sBt[(g + 1) * 64 + a] = (_Float16)(v.y * m);
      sBt[(g + 2) * 64 + a] = (_Float16)(v.z * m);
      sBt[(g + 3) * 64 + a] = (_Float16)(v.w * m);
    }
  }
  // subset tile (0/1 exact in f16): int4 loads, packed b64 LDS stores
  {
    const int4* s4 = (const int4*)(subsets + ((size_t)b * NS + s0) * NA);
    for (int i = tid; i < 128 * (NA / 4); i += 256) {  // 2048 chunks
      int r = i >> 4, c4 = i & 15;
      int4 v = s4[r * (NA / 4) + c4];
      int a = c4 * 4;
      v4h p;
      p[0] = (_Float16)((float)v.x * sMask[a + 0]);
      p[1] = (_Float16)((float)v.y * sMask[a + 1]);
      p[2] = (_Float16)((float)v.z * sMask[a + 2]);
      p[3] = (_Float16)((float)v.w * sMask[a + 3]);
      *(v4h*)(&sA[r * 64 + a]) = p;
    }
  }
  // W1 -> sW[d][g]; float4 over contiguous d, scatter 4 f16 stores
  {
    const float4* w4 = (const float4*)W1g;
    for (int i = tid; i < NG * (ND / 4); i += 256) {   // 4096 chunks
      int g = i >> 5, d4 = i & 31;
      float4 v = w4[g * (ND / 4) + d4];
      int d = d4 * 4;
      sW[(d + 0) * 128 + g] = (_Float16)v.x;
      sW[(d + 1) * 128 + g] = (_Float16)v.y;
      sW[(d + 2) * 128 + g] = (_Float16)v.z;
      sW[(d + 3) * 128 + g] = (_Float16)v.w;
    }
  }
  __syncthreads();

  if (tid < 128) {
    float s = 0.f;
#pragma unroll
    for (int a = 0; a < 64; ++a) s += (float)sA[tid * 64 + a];
    sSize[tid] = s + 1e-4f;
  }
  __syncthreads();

  const int lane = tid & 31;
  const int wv   = tid >> 5;
  const int rowBase = wv * 16;
  const int lm   = lane & 15;
  const int rb2  = rowBase + ((lane >> 4) << 3);  // row base for C elems (8 rows/lane)

  // ---- Stage 1: ssum = subs @ masked ; mean = ssum/size ; LayerNorm0 -> sX
  {
    v16h a0 = frag_a(sA, rowBase + lm, 64, 0,  lane);
    v16h a1 = frag_a(sA, rowBase + lm, 64, 32, lane);
    v8f acc[8];
#pragma unroll
    for (int t = 0; t < 8; ++t) {
      v8f c = {};
      c = wmma16(a0, frag_b(sBt, t * 16 + lm, 64, 0,  lane), c);
      c = wmma16(a1, frag_b(sBt, t * 16 + lm, 64, 32, lane), c);
      acc[t] = c;
    }
    float inv[8];
#pragma unroll
    for (int r = 0; r < 8; ++r) inv[r] = 1.0f / sSize[rb2 + r];
#pragma unroll
    for (int t = 0; t < 8; ++t)
#pragma unroll
      for (int r = 0; r < 8; ++r) acc[t][r] *= inv[r];
#pragma unroll
    for (int r = 0; r < 8; ++r) {
      float s = 0.f, q = 0.f;
#pragma unroll
      for (int t = 0; t < 8; ++t) { float v = acc[t][r]; s += v; q += v * v; }
      s = red16(s); q = red16(q);
      float mu  = s * (1.f / 128.f);
      float var = q * (1.f / 128.f) - mu * mu;
      float rs  = rsqrtf(var + LN_EPS);
#pragma unroll
      for (int t = 0; t < 8; ++t) {
        int col = t * 16 + lm;
        float y = (acc[t][r] - mu) * rs * sP[col] + sP[128 + col];
        sX[(rb2 + r) * 128 + col] = (_Float16)y;
      }
    }
  }
  __syncthreads();

  // ---- Stage 2: h = relu(x @ W1 + b1) -> sH
  {
    v16h xa[4];
#pragma unroll
    for (int k = 0; k < 4; ++k) xa[k] = frag_a(sX, rowBase + lm, 128, k * 32, lane);
#pragma unroll
    for (int t = 0; t < 8; ++t) {
      v8f c = {};
#pragma unroll
      for (int k = 0; k < 4; ++k)
        c = wmma16(xa[k], frag_b(sW, t * 16 + lm, 128, k * 32, lane), c);
      int col = t * 16 + lm;
      float bb = sP[2 * 128 + col];
#pragma unroll
      for (int r = 0; r < 8; ++r) {
        float y = fmaxf(c[r] + bb, 0.f);
        sH[(rb2 + r) * 128 + col] = (_Float16)y;
      }
    }
  }
  __syncthreads();
  // swap in W2 (transposed)
  {
    const float4* w4 = (const float4*)W2g;
    for (int i = tid; i < ND * (ND / 4); i += 256) {
      int g = i >> 5, d4 = i & 31;
      float4 v = w4[g * (ND / 4) + d4];
      int d = d4 * 4;
      sW[(d + 0) * 128 + g] = (_Float16)v.x;
      sW[(d + 1) * 128 + g] = (_Float16)v.y;
      sW[(d + 2) * 128 + g] = (_Float16)v.z;
      sW[(d + 3) * 128 + g] = (_Float16)v.w;
    }
  }
  __syncthreads();

  // ---- Stage 3: x = LN1(relu(h @ W2 + b2)) -> sX
  {
    v16h ha[4];
#pragma unroll
    for (int k = 0; k < 4; ++k) ha[k] = frag_a(sH, rowBase + lm, 128, k * 32, lane);
    v8f acc[8];
#pragma unroll
    for (int t = 0; t < 8; ++t) {
      v8f c = {};
#pragma unroll
      for (int k = 0; k < 4; ++k)
        c = wmma16(ha[k], frag_b(sW, t * 16 + lm, 128, k * 32, lane), c);
      int col = t * 16 + lm;
      float bb = sP[3 * 128 + col];
#pragma unroll
      for (int r = 0; r < 8; ++r) c[r] = fmaxf(c[r] + bb, 0.f);
      acc[t] = c;
    }
#pragma unroll
    for (int r = 0; r < 8; ++r) {
      float s = 0.f, q = 0.f;
#pragma unroll
      for (int t = 0; t < 8; ++t) { float v = acc[t][r]; s += v; q += v * v; }
      s = red16(s); q = red16(q);
      float mu  = s * (1.f / 128.f);
      float var = q * (1.f / 128.f) - mu * mu;
      float rs  = rsqrtf(var + LN_EPS);
#pragma unroll
      for (int t = 0; t < 8; ++t) {
        int col = t * 16 + lm;
        float y = (acc[t][r] - mu) * rs * sP[4 * 128 + col] + sP[5 * 128 + col];
        sX[(rb2 + r) * 128 + col] = (_Float16)y;
      }
    }
  }
  __syncthreads();

  // ---- Stage 4: scores = x @ Ws + bs  (one row per thread)
  if (tid < 128) {
    float s = sBs;
#pragma unroll 8
    for (int d = 0; d < 128; ++d) s += (float)sX[tid * 128 + d] * sP[6 * 128 + d];
    scores[(size_t)b * NS + s0 + tid] = s;
  }
}

// ---------------- Kernel 2: per-batch softmax over S=8192 ----------------

__global__ __launch_bounds__(256) void softmax_kernel(const float* __restrict__ scores,
                                                      float* __restrict__ probs) {
  __shared__ float red[256];
  const int b = blockIdx.x, tid = threadIdx.x;
  const float4* sc4 = (const float4*)(scores + (size_t)b * NS);
  float4* pb4 = (float4*)(probs + (size_t)b * NS);
  float4 vals[8];
  float mx = -3.4e38f;
#pragma unroll
  for (int i = 0; i < 8; ++i) {
    vals[i] = sc4[tid + i * 256];
    mx = fmaxf(mx, fmaxf(fmaxf(vals[i].x, vals[i].y), fmaxf(vals[i].z, vals[i].w)));
  }
  red[tid] = mx;
  __syncthreads();
  for (int s = 128; s > 0; s >>= 1) {
    if (tid < s) red[tid] = fmaxf(red[tid], red[tid + s]);
    __syncthreads();
  }
  mx = red[0];
  __syncthreads();
  float sum = 0.f;
#pragma unroll
  for (int i = 0; i < 8; ++i) {
    vals[i].x = __expf(vals[i].x - mx);
    vals[i].y = __expf(vals[i].y - mx);
    vals[i].z = __expf(vals[i].z - mx);
    vals[i].w = __expf(vals[i].w - mx);
    sum += (vals[i].x + vals[i].y) + (vals[i].z + vals[i].w);
  }
  red[tid] = sum;
  __syncthreads();
  for (int s = 128; s > 0; s >>= 1) {
    if (tid < s) red[tid] += red[tid + s];
    __syncthreads();
  }
  float inv = 1.f / red[0];
#pragma unroll
  for (int i = 0; i < 8; ++i) {
    float4 o;
    o.x = vals[i].x * inv; o.y = vals[i].y * inv;
    o.z = vals[i].z * inv; o.w = vals[i].w * inv;
    pb4[tid + i * 256] = o;
  }
}

// ---------------- Kernel 3: per-batch LDS histogram scatter ----------------
// one block per batch: S*M = 131072 peaks -> 512 bins via ds_add_f32, then plain stores.

__global__ __launch_bounds__(256) void scatter_kernel(const float* __restrict__ peaks,
                                                      const float* __restrict__ probs,
                                                      float* __restrict__ spect) {
  __shared__ float bins[SPECT_BIN_N];
  const int b = blockIdx.x, tid = threadIdx.x;
  bins[tid] = 0.f;
  bins[tid + 256] = 0.f;
  __syncthreads();
  const float2* pk = (const float2*)(peaks + (size_t)b * NS * NM * 2);
  const float* pr = probs + (size_t)b * NS;
  for (int i = 0; i < (NS * NM) / 256; ++i) {
    int p = tid + i * 256;     // peak index within batch
    int s = p >> 4;            // subset
    float2 mi = pk[p];         // (mass, intensity) as one b64 load
    int bin = __float2int_rn(mi.x);
    bin = bin < 0 ? 0 : (bin > SPECT_BIN_N - 1 ? SPECT_BIN_N - 1 : bin);
    unsafeAtomicAdd(&bins[bin], mi.y * pr[s]);
  }
  __syncthreads();
  spect[(size_t)b * SPECT_BIN_N + tid]       = bins[tid];
  spect[(size_t)b * SPECT_BIN_N + 256 + tid] = bins[tid + 256];
}

// ---------------- launch ----------------

extern "C" void kernel_launch(void* const* d_in, const int* in_sizes, int n_in,
                              void* d_out, int out_size, void* d_ws, size_t ws_size,
                              hipStream_t stream) {
  const float* feats   = (const float*)d_in[0];   // (B,A,G)
  const float* mask    = (const float*)d_in[1];   // (B,A)
  // d_in[2] vert_element_oh, d_in[3] adj_oh: unused by the reference
  const int*   subsets = (const int*)d_in[4];     // (B,S,A)
  const float* peaks   = (const float*)d_in[5];   // (B,S,M,2)
  const float* ln0w = (const float*)d_in[6];
  const float* ln0b = (const float*)d_in[7];
  const float* W1   = (const float*)d_in[8];
  const float* b1   = (const float*)d_in[9];
  const float* W2   = (const float*)d_in[10];
  const float* b2   = (const float*)d_in[11];
  const float* ln1w = (const float*)d_in[12];
  const float* ln1b = (const float*)d_in[13];
  const float* Ws   = (const float*)d_in[14];
  const float* bs   = (const float*)d_in[15];

  float* out   = (float*)d_out;
  float* spect = out;                            // (B, 512)
  float* probs = out + (size_t)NB * SPECT_BIN_N; // (B, S)
  float* scores = (float*)d_ws;                  // (B, S) scratch

  dim3 g1(NS / 128, NB);
  fused_scorer<<<g1, 256, 0, stream>>>(feats, mask, subsets,
                                       ln0w, ln0b, W1, b1, W2, b2,
                                       ln1w, ln1b, Ws, bs, scores);
  softmax_kernel<<<NB, 256, 0, stream>>>(scores, probs);
  scatter_kernel<<<NB, 256, 0, stream>>>(peaks, probs, spect);
}
